// Low_Rank_RNN_76020921140338
// MI455X (gfx1250) — compile-verified, matrix operand order checked by reading, not run on hardware
//
#include <hip/hip_runtime.h>
#include <hip/hip_bf16.h>
#include <math.h>

// Problem constants (from the reference)
constexpr int HID = 512;
constexpr int IN  = 4;
constexpr int OUT = 2;
constexpr int B   = 128;
constexpr int T   = 512;
constexpr float NOISE_STD = 0.05f;
constexpr float TAU       = 0.2f;

typedef __attribute__((ext_vector_type(16))) __bf16 v16bf;
typedef __attribute__((ext_vector_type(8)))  float  v8f;

// Branchless tanh: tanh(x) = (e^{2x}-1)/(e^{2x}+1), e^{2x} = 2^{x * 2log2(e)}.
// Lowers to v_med3_num_f32 (fused clamp) + v_exp_f32 + v_rcp_f32 + pk ops,
// with NO exec-mask branches, so surrounding loads stay vectorized
// (global_load_b128) and the scan's serial critical path stays short.
__device__ __forceinline__ float fast_tanh(float x) {
    const float TWO_LOG2E = 2.8853900817779268f;
    float xc = fminf(fmaxf(x, -9.0f), 9.0f);
    float t  = __builtin_amdgcn_exp2f(xc * TWO_LOG2E);
    return (t - 1.0f) * __builtin_amdgcn_rcpf(t + 1.0f);
}

// ---------------------------------------------------------------------------
// Kernel 1: sequential Euler scan. One block per batch element, one thread per
// hidden unit (512 threads = 16 wave32 waves). Writes trajectories + x_last.
//
// Per-step cross-wave reduction uses ONE barrier via double-buffered LDS
// partials: each wave's lane0 writes a float2 partial, one __syncthreads,
// then every thread sums all 16 partials itself (LDS broadcast reads).
// Double buffering on (t&1) makes the single-barrier scheme race-free.
// ---------------------------------------------------------------------------
__global__ __launch_bounds__(HID)
void lrrnn_scan_kernel(const float* __restrict__ u,
                       const float* __restrict__ x0,
                       const float* __restrict__ noise,
                       const float* __restrict__ M,
                       const float* __restrict__ N,
                       const float* __restrict__ w_in_w,
                       const float* __restrict__ w_in_b,
                       float* __restrict__ x_last,
                       float* __restrict__ traj) {
    const int b    = blockIdx.x;
    const int h    = threadIdx.x;
    const int lane = h & 31;
    const int wave = h >> 5;

    __shared__ float2 red[2][16];

    // Per-thread constants (stay in VGPRs for all 512 steps)
    const float m0 = M[h * 2 + 0];
    const float m1 = M[h * 2 + 1];
    const float n0 = N[h * 2 + 0];
    const float n1 = N[h * 2 + 1];
    const float wi0 = w_in_w[h * 4 + 0];
    const float wi1 = w_in_w[h * 4 + 1];
    const float wi2 = w_in_w[h * 4 + 2];
    const float wi3 = w_in_w[h * 4 + 3];
    const float bi  = w_in_b[h];

    float x = x0[(size_t)b * HID + h];

    for (int t = 0; t < T; ++t) {
        // Issue the loads that do NOT depend on the reduction first, so their
        // latency overlaps the shuffle/LDS reduction below.
        const float nval = noise[((size_t)t * B + b) * HID + h];
        const float4 uv  = *(const float4*)(u + ((size_t)b * T + t) * IN); // uniform -> scalar path
        if (t + 1 < T) {
            // gfx1250: global_prefetch_b8 for next step's noise row
            __builtin_prefetch(&noise[((size_t)(t + 1) * B + b) * HID + h], 0, 1);
        }

        const float r = fast_tanh(x);

        // kappa_k = sum_h r[h] * N[h,k]  (two simultaneous wave reductions)
        float p0 = r * n0;
        float p1 = r * n1;
        #pragma unroll
        for (int off = 16; off >= 1; off >>= 1) {
            p0 += __shfl_xor(p0, off, 32);
            p1 += __shfl_xor(p1, off, 32);
        }
        if (lane == 0) { red[t & 1][wave] = make_float2(p0, p1); }
        __syncthreads();

        // Every thread reduces the 16 wave partials itself (broadcast reads).
        float k0 = 0.0f, k1 = 0.0f;
        #pragma unroll
        for (int wv = 0; wv < 16; ++wv) {
            const float2 pr = red[t & 1][wv];
            k0 += pr.x;
            k1 += pr.y;
        }

        const float drive = bi + uv.x * wi0 + uv.y * wi1 + uv.z * wi2 + uv.w * wi3;
        const float hidden_W = (k0 * m0 + k1 * m1) * (1.0f / (float)HID);

        x = x + NOISE_STD * nval + TAU * (-x + hidden_W + drive);

        traj[((size_t)b * T + t) * HID + h] = x;
    }

    x_last[(size_t)b * HID + h] = x;
}

// ---------------------------------------------------------------------------
// Kernel 2: output projection  out[row, o] = sum_k tanh(traj[row, k]) * w_out[o, k] + b[o]
// as a WMMA bf16 GEMM: rows tiled by 16, K=512 chained as 16 x (16x16x32),
// N padded 2 -> 16 with zero columns. One wave per row tile.
// ---------------------------------------------------------------------------
__global__ __launch_bounds__(256)
void lrrnn_outproj_kernel(const float* __restrict__ traj,
                          const float* __restrict__ w_out_w,
                          const float* __restrict__ w_out_b,
                          float* __restrict__ out) {
    const int lane = threadIdx.x & 31;
    const int wave = threadIdx.x >> 5;
    const int tile = blockIdx.x * (blockDim.x >> 5) + wave; // 16-row tile id
    const int row0 = tile * 16;
    const int m = lane & 15;   // A row within tile / D column index
    const int g = lane >> 4;   // lane group (k-half select)

    v8f acc = {};

    #pragma unroll 4
    for (int kc = 0; kc < HID / 32; ++kc) {
        const int kbase = kc * 32;

        // ---- A operand: 16x32 bf16 tile of tanh(traj) -------------------
        // ISA 16-bit A layout: lane group g, element j  <->  k = (j<8 ? j : j+8) + 8*g
        const float* arow = traj + (size_t)(row0 + m) * HID + kbase + g * 8;
        const float4 a0 = *(const float4*)(arow + 0);
        const float4 a1 = *(const float4*)(arow + 4);
        const float4 a2 = *(const float4*)(arow + 16);
        const float4 a3 = *(const float4*)(arow + 20);

        v16bf Av;
        Av[0]  = (__bf16)fast_tanh(a0.x);
        Av[1]  = (__bf16)fast_tanh(a0.y);
        Av[2]  = (__bf16)fast_tanh(a0.z);
        Av[3]  = (__bf16)fast_tanh(a0.w);
        Av[4]  = (__bf16)fast_tanh(a1.x);
        Av[5]  = (__bf16)fast_tanh(a1.y);
        Av[6]  = (__bf16)fast_tanh(a1.z);
        Av[7]  = (__bf16)fast_tanh(a1.w);
        Av[8]  = (__bf16)fast_tanh(a2.x);
        Av[9]  = (__bf16)fast_tanh(a2.y);
        Av[10] = (__bf16)fast_tanh(a2.z);
        Av[11] = (__bf16)fast_tanh(a2.w);
        Av[12] = (__bf16)fast_tanh(a3.x);
        Av[13] = (__bf16)fast_tanh(a3.y);
        Av[14] = (__bf16)fast_tanh(a3.z);
        Av[15] = (__bf16)fast_tanh(a3.w);

        // ---- B operand: 32x16 bf16, column n = lane&15, zero for n >= OUT
        v16bf Bv;
        #pragma unroll
        for (int j = 0; j < 16; ++j) Bv[j] = (__bf16)0.0f;
        if (m < OUT) {
            const float* wrow = w_out_w + m * HID + kbase + g * 8;
            const float4 b0 = *(const float4*)(wrow + 0);
            const float4 b1 = *(const float4*)(wrow + 4);
            const float4 b2 = *(const float4*)(wrow + 16);
            const float4 b3 = *(const float4*)(wrow + 20);
            Bv[0]  = (__bf16)b0.x;  Bv[1]  = (__bf16)b0.y;
            Bv[2]  = (__bf16)b0.z;  Bv[3]  = (__bf16)b0.w;
            Bv[4]  = (__bf16)b1.x;  Bv[5]  = (__bf16)b1.y;
            Bv[6]  = (__bf16)b1.z;  Bv[7]  = (__bf16)b1.w;
            Bv[8]  = (__bf16)b2.x;  Bv[9]  = (__bf16)b2.y;
            Bv[10] = (__bf16)b2.z;  Bv[11] = (__bf16)b2.w;
            Bv[12] = (__bf16)b3.x;  Bv[13] = (__bf16)b3.y;
            Bv[14] = (__bf16)b3.z;  Bv[15] = (__bf16)b3.w;
        }

        // D = A x B + C ; accumulates the K=512 contraction in f32
        acc = __builtin_amdgcn_wmma_f32_16x16x32_bf16(
            /*neg_a=*/false, Av, /*neg_b=*/false, Bv,
            /*c_mod=*/(short)0, acc, /*reuse_a=*/false, /*reuse_b=*/false);
    }

    // C/D layout: N = lane&15 ; VGPR v holds M = v + 8*(lane>>4)
    if (m < OUT) {
        const float bias = w_out_b[m];
        #pragma unroll
        for (int v = 0; v < 8; ++v) {
            const int row = row0 + v + g * 8;
            out[(size_t)row * OUT + m] = acc[v] + bias;
        }
    }
}

// ---------------------------------------------------------------------------
// Launch. d_out layout (reference return order): output[B*T*OUT] | x_last[B*HID]
// | trajectories[B*T*HID]. Kernel 2 reads trajectories written by kernel 1
// (same stream => ordered).
// ---------------------------------------------------------------------------
extern "C" void kernel_launch(void* const* d_in, const int* in_sizes, int n_in,
                              void* d_out, int out_size, void* d_ws, size_t ws_size,
                              hipStream_t stream) {
    const float* u       = (const float*)d_in[0];
    const float* x0      = (const float*)d_in[1];
    const float* noise   = (const float*)d_in[2];
    const float* M       = (const float*)d_in[3];
    const float* N       = (const float*)d_in[4];
    const float* w_in_w  = (const float*)d_in[5];
    const float* w_in_b  = (const float*)d_in[6];
    const float* w_out_w = (const float*)d_in[7];
    const float* w_out_b = (const float*)d_in[8];

    float* out_output = (float*)d_out;
    float* out_xlast  = out_output + (size_t)B * T * OUT;
    float* out_traj   = out_xlast + (size_t)B * HID;

    lrrnn_scan_kernel<<<B, HID, 0, stream>>>(u, x0, noise, M, N, w_in_w, w_in_b,
                                             out_xlast, out_traj);

    const int tiles = (B * T) / 16;          // 4096 row tiles
    const int wavesPerBlock = 8;             // 256 threads
    lrrnn_outproj_kernel<<<tiles / wavesPerBlock, 32 * wavesPerBlock, 0, stream>>>(
        out_traj, w_out_w, w_out_b, out_output);
}
